// MultiHeadAttention_10222022165355
// MI455X (gfx1250) — matrix operand order repaired and checked by reading, not compiled, not run on hardware
//
#include <hip/hip_runtime.h>

// ---------------------------------------------------------------------------
// Fused MHA for MI455X (gfx1250, wave32, WMMA).
// B=64,S=8,A=128,E=256,H=4,D=64 -> 65536 query rows, 2 keys per query.
// All four GEMMs run on v_wmma_f32_16x16x32_bf16 with f32 accumulation.
// Tile staging uses GLOBAL_LOAD_ASYNC_TO_LDS (ASYNCcnt) when available.
// ---------------------------------------------------------------------------

typedef __bf16 v16bf __attribute__((ext_vector_type(16)));
typedef float  v8f   __attribute__((ext_vector_type(8)));
typedef int    v4i   __attribute__((ext_vector_type(4)));

#define E_DIM   256
#define M_TILE  64
#define KSTEPS  8      // 256 / 32
#define THREADS 256

#if __has_builtin(__builtin_amdgcn_global_load_async_to_lds_b128) && \
    __has_builtin(__builtin_amdgcn_s_wait_asynccnt)
#define USE_ASYNC_LDS 1
typedef __attribute__((address_space(1))) v4i* gptr_v4i;  // global (AS1)
typedef __attribute__((address_space(3))) v4i* lptr_v4i;  // LDS (AS3)
#else
#define USE_ASYNC_LDS 0
#endif

// round-to-nearest-even f32 -> bf16 (bit pattern)
static __device__ __forceinline__ unsigned short f2bf(float f) {
  unsigned u = __float_as_uint(f);
  u += 0x7FFFu + ((u >> 16) & 1u);
  return (unsigned short)(u >> 16);
}
// unpack a packed bf16 pair (one shift + one mask)
static __device__ __forceinline__ float bf_lo(unsigned u) {
  return __uint_as_float(u << 16);
}
static __device__ __forceinline__ float bf_hi(unsigned u) {
  return __uint_as_float(u & 0xFFFF0000u);
}

// --- A-fragment swizzle for V_WMMA_*_16X16X32 (16-bit A, 16x32 MxK) --------
// lane L (0..15): row M=L, K = {0..7,16..23}; lane L+16: K = {8..15,24..31}
// element e = 2*p + (k&1); p = (k32>=16 ? 4:0) + (k32&7)/2
// LDS layout: idx = (((row*8 + ks)*2 + half)*16 + e)   (bf16 elements)
static __device__ __forceinline__ void store_pair_swz(unsigned short* dst,
                                                      int row, int k,
                                                      float f0, float f1) {
  int ks   = k >> 5;
  int k32  = k & 31;
  int half = (k32 >> 3) & 1;
  int p4   = ((k32 & 16) >> 2) + ((k32 & 7) >> 1);
  int idx  = (((row << 3) + ks) * 2 + half) * 16 + (p4 << 1);
  ((unsigned*)dst)[idx >> 1] =
      (unsigned)f2bf(f0) | ((unsigned)f2bf(f1) << 16);
}

// Stage a 64x256 f32 tile from global into LDS as swizzled bf16 A-fragments.
// Async path: HBM -> LDS (f32, no VGPRs, ASYNCcnt) -> convert/swizzle.
// Fallback:   HBM -> VGPR (float4) -> convert/swizzle.
static __device__ __forceinline__ void stage_tile(const float* __restrict__ src,
                                                  float* Xf,  // LDS scratch 64KB
                                                  unsigned short* dst) {
  const int t = threadIdx.x;
#if USE_ASYNC_LDS
#pragma unroll
  for (int i = 0; i < 16; ++i) {
    v4i* g = (v4i*)(src + i * 1024 + t * 4);  // drops const, stays generic
    v4i* l = (v4i*)(Xf + i * 1024 + t * 4);
    __builtin_amdgcn_global_load_async_to_lds_b128(
        (gptr_v4i)g, (lptr_v4i)l, 0, 0);
  }
  __builtin_amdgcn_s_wait_asynccnt(0);
  __syncthreads();
  const float4* s4 = (const float4*)Xf;
#else
  (void)Xf;
  const float4* s4 = (const float4*)src;
#endif
#pragma unroll
  for (int i = 0; i < 16; ++i) {
    float4 v = s4[i * 256 + t];
    int f   = i * 1024 + t * 4;  // flat float index in the 64x256 tile
    int row = f >> 8;
    int k   = f & 255;
    store_pair_swz(dst, row, k,     v.x, v.y);
    store_pair_swz(dst, row, k + 2, v.z, v.w);
  }
}

// --- GEMM tile: [64 x 256] = A(LDS, swizzled bf16) x W^T(global, frag-linear)
// Wave w: msub = w/2 (16-row strip), nhalf = w%2 (128-col strip -> 8 subtiles).
// Wp frag-linear: element(((ks*16+nt)*32 + lane)*16 + j) = W[nt*16+lane%16][ks*32+(j>>3)*16+(lane>>4)*8+(j&7)]
template <bool TO_LDS>
static __device__ __forceinline__ void gemm_tile(
    const unsigned short* As,                // LDS, 64x256 swizzled bf16
    const unsigned short* __restrict__ Wp,   // global, 65536 bf16 frag-linear
    const float* __restrict__ bias,          // global f32 [256]
    unsigned short* outT,                    // LDS row-major [64][256] bf16
    float* __restrict__ outG)                // global f32 tile base
{
  const int lane  = threadIdx.x & 31;
  const int wave  = threadIdx.x >> 5;
  const int msub  = wave >> 1;
  const int nhalf = wave & 1;

  v8f acc[8];
#pragma unroll
  for (int j = 0; j < 8; ++j) acc[j] = (v8f)0.0f;

#pragma unroll
  for (int ks = 0; ks < KSTEPS; ++ks) {
    const v16bf a = *(const v16bf*)(As +
        ((((msub * 16 + (lane & 15)) * 8 + ks) * 2 + (lane >> 4)) * 16));
#pragma unroll
    for (int j = 0; j < 8; ++j) {
      const int nt  = nhalf * 8 + j;
      const v16bf b = *(const v16bf*)(Wp + (((ks * 16 + nt) * 32 + lane) * 16));
      acc[j] = __builtin_amdgcn_wmma_f32_16x16x32_bf16(
          false, a, false, b, (short)0, acc[j], false, false);
    }
  }

  // C/D layout: lanes 0-15 -> M=i, lanes 16-31 -> M=i+8 ; N = lane&15
  const int rbase = msub * 16 + ((lane >> 4) << 3);
  const int cl    = lane & 15;
#pragma unroll
  for (int j = 0; j < 8; ++j) {
    const int col = (nhalf * 8 + j) * 16 + cl;
    const float bv = bias[col];
#pragma unroll
    for (int i = 0; i < 8; ++i) {
      const float v = acc[j][i] + bv;
      const int row = rbase + i;
      if constexpr (TO_LDS) outT[row * E_DIM + col] = f2bf(v);
      else                  outG[(size_t)row * E_DIM + col] = v;
    }
  }
}

// --- attention over 2 keys; writes concat rows into swizzled LDS ------------
// Vectorized: all Q/K/V reads are uint4 (8 bf16) -> ds_load_b128.
static __device__ __forceinline__ void attention_half(
    const unsigned short* Qt,    // [64][256] bf16, query rows of tile
    const unsigned short* Kt,    // [64][256] bf16, this half's keys
    const unsigned short* Vt,    // [64][256] bf16
    int hf, unsigned short* Cs)  // concat, swizzled A layout
{
  const int t = threadIdx.x;
  if (t < 128) {
    const int rl   = t >> 2;        // local query row within half (0..31)
    const int h    = t & 3;         // head
    const int qrow = hf * 32 + rl;  // row within 64-query tile
    const uint4* q4  = (const uint4*)(Qt + qrow * E_DIM + h * 64);
    const uint4* k04 = (const uint4*)(Kt + (2 * rl) * E_DIM + h * 64);
    const uint4* k14 = (const uint4*)(Kt + (2 * rl + 1) * E_DIM + h * 64);
    const uint4* v04 = (const uint4*)(Vt + (2 * rl) * E_DIM + h * 64);
    const uint4* v14 = (const uint4*)(Vt + (2 * rl + 1) * E_DIM + h * 64);

    float s0 = 0.f, s1 = 0.f;
#pragma unroll
    for (int i = 0; i < 8; ++i) {  // 8 x (8 bf16) = 64 dims
      const uint4 qa = q4[i], ka = k04[i], kb = k14[i];
      const unsigned qw[4] = {qa.x, qa.y, qa.z, qa.w};
      const unsigned aw[4] = {ka.x, ka.y, ka.z, ka.w};
      const unsigned bw[4] = {kb.x, kb.y, kb.z, kb.w};
#pragma unroll
      for (int j = 0; j < 4; ++j) {
        s0 = fmaf(bf_lo(qw[j]), bf_lo(aw[j]), s0);
        s0 = fmaf(bf_hi(qw[j]), bf_hi(aw[j]), s0);
        s1 = fmaf(bf_lo(qw[j]), bf_lo(bw[j]), s1);
        s1 = fmaf(bf_hi(qw[j]), bf_hi(bw[j]), s1);
      }
    }
    s0 *= 0.125f;  // 1/sqrt(64)
    s1 *= 0.125f;
    const float m   = fmaxf(s0, s1);
    const float e0  = __expf(s0 - m);
    const float e1  = __expf(s1 - m);
    const float inv = 1.f / (e0 + e1);
    const float w0 = e0 * inv, w1 = e1 * inv;

#pragma unroll
    for (int i = 0; i < 8; ++i) {
      const uint4 va = v04[i], vb = v14[i];
      const unsigned aw[4] = {va.x, va.y, va.z, va.w};
      const unsigned bw[4] = {vb.x, vb.y, vb.z, vb.w};
#pragma unroll
      for (int j = 0; j < 4; ++j) {
        const float a0 = w0 * bf_lo(aw[j]) + w1 * bf_lo(bw[j]);
        const float a1 = w0 * bf_hi(aw[j]) + w1 * bf_hi(bw[j]);
        store_pair_swz(Cs, qrow, h * 64 + i * 8 + j * 2, a0, a1);
      }
    }
  }
}

// --- weight pre-pass: f32 [n][k] row-major  ->  bf16 frag-linear ------------
__global__ __launch_bounds__(256) void convert_weights(
    const float* __restrict__ W, unsigned short* __restrict__ Wp) {
  const int idx = blockIdx.x * 256 + threadIdx.x;  // 0..65535
  const int j  = idx & 15;
  const int kl = (idx >> 4) & 31;
  const int nt = (idx >> 9) & 15;
  const int ks = idx >> 13;
  Wp[idx] = f2bf(W[(nt * 16 + (kl & 15)) * E_DIM + ks * 32 + ((j & 8) << 1) + ((kl >> 4) << 3) + (j & 7)]);
}

// --- fused main kernel ------------------------------------------------------
__global__ __launch_bounds__(THREADS) void mha_fused(
    const float* __restrict__ p,     // [65536][256]
    const float* __restrict__ c,     // [131072][256]
    const unsigned short* __restrict__ Wqp, const float* __restrict__ bq,
    const unsigned short* __restrict__ Wkp, const float* __restrict__ bk,
    const unsigned short* __restrict__ Wvp, const float* __restrict__ bv,
    const unsigned short* __restrict__ Wop, const float* __restrict__ bo,
    float* __restrict__ out)         // [65536][256]
{
  // one carved block: 5 x 32KB = 160KB -> 2 workgroups per WGP (320KB pool)
  __shared__ __align__(32) unsigned short smem[5 * M_TILE * E_DIM];
  unsigned short* Xs = smem;            // staging, swizzled A  (32KB)
  unsigned short* Qt = smem + 16384;    // Q rows bf16          (32KB)
  unsigned short* Kt = smem + 32768;    // K half bf16          (32KB)
  unsigned short* Vt = smem + 49152;    // V half bf16          (32KB)
  unsigned short* Cs = smem + 65536;    // concat, swizzled A   (32KB)
  float* Xf = (float*)(smem + 32768);   // f32 async staging, aliases Kt+Vt
                                        // (dead whenever staging runs)

  const size_t m0 = (size_t)blockIdx.x * M_TILE;

  // Q = p @ Wq^T + bq
  stage_tile(p + m0 * E_DIM, Xf, Xs);
  __syncthreads();
  gemm_tile<true>(Xs, Wqp, bq, Qt, nullptr);
  __syncthreads();

  // two halves of the 128 key/value rows belonging to this query tile
  for (int hf = 0; hf < 2; ++hf) {
    stage_tile(c + (2 * m0 + (size_t)hf * 64) * E_DIM, Xf, Xs);
    __syncthreads();
    gemm_tile<true>(Xs, Wkp, bk, Kt, nullptr);
    gemm_tile<true>(Xs, Wvp, bv, Vt, nullptr);
    __syncthreads();
    attention_half(Qt, Kt, Vt, hf, Cs);
    __syncthreads();
  }

  // out = concat @ Wo^T + bo
  gemm_tile<false>(Cs, Wop, bo, nullptr, out + m0 * E_DIM);
}

// ---------------------------------------------------------------------------
extern "C" void kernel_launch(void* const* d_in, const int* in_sizes, int n_in,
                              void* d_out, int out_size, void* d_ws, size_t ws_size,
                              hipStream_t stream) {
  (void)in_sizes; (void)n_in; (void)out_size; (void)ws_size;
  const float* p  = (const float*)d_in[0];
  const float* c  = (const float*)d_in[1];
  const float* Wq = (const float*)d_in[2];
  const float* bq = (const float*)d_in[3];
  const float* Wk = (const float*)d_in[4];
  const float* bk = (const float*)d_in[5];
  const float* Wv = (const float*)d_in[6];
  const float* bv = (const float*)d_in[7];
  const float* Wo = (const float*)d_in[8];
  const float* bo = (const float*)d_in[9];
  float* out = (float*)d_out;

  unsigned short* Wqp = (unsigned short*)d_ws;  // 4 x 65536 bf16 = 512KB
  unsigned short* Wkp = Wqp + 65536;
  unsigned short* Wvp = Wkp + 65536;
  unsigned short* Wop = Wvp + 65536;

  convert_weights<<<256, 256, 0, stream>>>(Wq, Wqp);
  convert_weights<<<256, 256, 0, stream>>>(Wk, Wkp);
  convert_weights<<<256, 256, 0, stream>>>(Wv, Wvp);
  convert_weights<<<256, 256, 0, stream>>>(Wo, Wop);

  mha_fused<<<1024, THREADS, 0, stream>>>(p, c, Wqp, bq, Wkp, bk,
                                          Wvp, bv, Wop, bo, out);
}
